// DPRNN_12154757447884
// MI455X (gfx1250) — matrix operand order repaired and checked by reading, not compiled, not run on hardware
//
#include <hip/hip_runtime.h>

// ---------------------------------------------------------------------------
// Problem constants (from the reference)
// ---------------------------------------------------------------------------
#define Cc    64
#define Fdim  128
#define Tdim  250
#define Kw    8
#define Tn    243          // Tdim - Kw + 1
#define Nrow  256          // B * Fdim
#define Bb    2
#define Hh    32
#define H2    64
#define H4    128
#define Tpad  256          // padded t-stride for bf16 input

typedef __bf16 v16bf __attribute__((ext_vector_type(16)));
typedef float  v8f   __attribute__((ext_vector_type(8)));

__device__ __forceinline__ unsigned short f2bf(float f) {
    unsigned int u = __float_as_uint(f);
    unsigned int r = u + 0x7FFFu + ((u >> 16) & 1u);
    return (unsigned short)(r >> 16);
}
__device__ __forceinline__ float bf2f(unsigned short h) {
    return __uint_as_float(((unsigned int)h) << 16);
}

__device__ __forceinline__ v8f wmma_bf16(v16bf a, v16bf b, v8f c) {
    return __builtin_amdgcn_wmma_f32_16x16x32_bf16(
        /*neg_a=*/false, a, /*neg_b=*/false, b,
        /*c_mod=*/(short)0, c, /*reuse_a=*/false, /*reuse_b=*/false);
}

// ---------------------------------------------------------------------------
// Stage 0: per-batch mean / rstd  (two-stage deterministic reduction)
// ---------------------------------------------------------------------------
__global__ __launch_bounds__(256)
void stats_part_kernel(const float* __restrict__ x, float* __restrict__ part) {
    int blk = blockIdx.x;          // 256 blocks: 128 per batch
    int bat = blk >> 7;
    int sub = blk & 127;
    const long n = (long)Cc * Fdim * Tdim;   // 2,048,000
    const float* xb = x + bat * n;
    float s = 0.f, s2 = 0.f;
    for (long i = (long)sub * 256 + threadIdx.x; i < n; i += 128L * 256) {
        float v = xb[i]; s += v; s2 += v * v;
    }
    __shared__ float sh[512];
    sh[threadIdx.x] = s; sh[256 + threadIdx.x] = s2;
    __syncthreads();
    for (int o = 128; o; o >>= 1) {
        if ((int)threadIdx.x < o) {
            sh[threadIdx.x]       += sh[threadIdx.x + o];
            sh[256 + threadIdx.x] += sh[256 + threadIdx.x + o];
        }
        __syncthreads();
    }
    if (threadIdx.x == 0) { part[blk * 2] = sh[0]; part[blk * 2 + 1] = sh[256]; }
}

__global__ void stats_final_kernel(const float* __restrict__ part, float* __restrict__ stats) {
    int bat = threadIdx.x;
    if (bat < Bb) {
        float s = 0.f, s2 = 0.f;
        for (int i = 0; i < 128; ++i) {
            s  += part[(bat * 128 + i) * 2];
            s2 += part[(bat * 128 + i) * 2 + 1];
        }
        const float n = (float)((long)Cc * Fdim * Tdim);
        float mean = s / n;
        float var  = s2 / n - mean * mean;
        stats[bat * 2]     = mean;
        stats[bat * 2 + 1] = rsqrtf(var + 1e-5f);
    }
}

// ---------------------------------------------------------------------------
// Stage 1: normalize x and split into bf16 hi/lo, layout [n][c][Tpad]
// ---------------------------------------------------------------------------
__global__ __launch_bounds__(256)
void pack_x_kernel(const float* __restrict__ x, const float* __restrict__ gamma,
                   const float* __restrict__ beta, const float* __restrict__ stats,
                   unsigned short* __restrict__ xhi, unsigned short* __restrict__ xlo) {
    int idx = blockIdx.x * 256 + threadIdx.x;
    const int total = Nrow * Cc * Tdim;
    if (idx >= total) return;
    int t = idx % Tdim;
    int c = (idx / Tdim) % Cc;
    int n = idx / (Tdim * Cc);
    int bat = n >> 7, f = n & 127;
    float v = x[(((long)(bat * Cc + c)) * Fdim + f) * Tdim + t];
    v = (v - stats[2 * bat]) * stats[2 * bat + 1] * gamma[c] + beta[c];
    unsigned short hi = f2bf(v);
    long o = ((long)n * Cc + c) * Tpad + t;
    xhi[o] = hi;
    xlo[o] = f2bf(v - bf2f(hi));
}

// ---------------------------------------------------------------------------
// Stage 2: pre-swizzle weights into WMMA B-fragment layout (bf16 hi/lo)
//   B-frag (32x16 bf16): lane l -> col = l&15, k = ((l&16)?16:0) + elem i
//   packed index: ((kt*16 + ct)*32 + l)*16 + i
// pW0: Ktot=512 (kt 0..15), 256 cols (dir0 j | dir1 j).  pWr: Ktot=64, x3 layers.
// Also transpose wct -> wt[c][m][i] for the conv.
// ---------------------------------------------------------------------------
__global__ __launch_bounds__(256)
void pack_w_kernel(const float* __restrict__ W0, const float* __restrict__ Wr,
                   const float* __restrict__ wct,
                   unsigned short* __restrict__ pW0h, unsigned short* __restrict__ pW0l,
                   unsigned short* __restrict__ pWrh, unsigned short* __restrict__ pWrl,
                   float* __restrict__ wt) {
    int idx = blockIdx.x * 256 + threadIdx.x;
    if (idx < 131072) {                       // W0: 16kt * 16ct * 32 * 16
        int p = idx;
        int i = p & 15, l = (p >> 4) & 31, ct = (p >> 9) & 15, kt = p >> 13;
        int k   = kt * 32 + ((l & 16) ? 16 : 0) + i;
        int col = ct * 16 + (l & 15);
        int dir = col >> 7, j = col & 127;
        float v = W0[((long)(dir * 512 + k)) * H4 + j];
        unsigned short h = f2bf(v);
        pW0h[p] = h; pW0l[p] = f2bf(v - bf2f(h));
    } else if (idx < 131072 + 49152) {        // Wr: 3 layers * 2kt * 16ct * 32 * 16
        int p = idx - 131072;
        int lay = p / 16384, p2 = p - lay * 16384;
        int i = p2 & 15, l = (p2 >> 4) & 31, ct = (p2 >> 9) & 15, kt = p2 >> 13;
        int k   = kt * 32 + ((l & 16) ? 16 : 0) + i;
        int col = ct * 16 + (l & 15);
        int dir = col >> 7, j = col & 127;
        float v = Wr[((long)((lay * 2 + dir) * H2 + k)) * H4 + j];
        unsigned short h = f2bf(v);
        pWrh[p] = h; pWrl[p] = f2bf(v - bf2f(h));
    } else if (idx < 131072 + 49152 + 32768) { // wct transpose: (i,c,m) -> wt[c][m][i]
        int p = idx - 131072 - 49152;
        int i2 = p & 63, m = (p >> 6) & 7, c = p >> 9;
        wt[p] = wct[((long)i2 * Cc + c) * Kw + m];
    }
}

// ---------------------------------------------------------------------------
// Stage 3: tiled WMMA GEMM, bf16x3 split accumulate in f32.
//   Block = (t, 16-row n-tile). 8 waves, each owns 2 of 16 column tiles.
//   WINDOWED: A row = concat_c xu[n][c][t..t+7]   (layer 0, Ktot=512)
//   else    : A row = hcat[t][n][0..63]            (layers 1..3, Ktot=64)
//   Output: U[t][n][256] f32  (cols 0..127 dir0, 128..255 dir1)
// ---------------------------------------------------------------------------
template<int NKT, bool WINDOWED>
__global__ __launch_bounds__(256)
void gemm_kernel(const unsigned short* __restrict__ Ahi,
                 const unsigned short* __restrict__ Alo,
                 const unsigned short* __restrict__ Bhi,
                 const unsigned short* __restrict__ Blo,
                 float* __restrict__ U) {
    __shared__ v16bf sAh[NKT * 32];
    __shared__ v16bf sAl[NKT * 32];
    unsigned short* sAhp = (unsigned short*)sAh;
    unsigned short* sAlp = (unsigned short*)sAl;

    const int t  = blockIdx.x >> 4;         // 0..242
    const int n0 = (blockIdx.x & 15) * 16;
    const int tid = threadIdx.x;

    // Cooperative A-tile staging in A-fragment layout:
    // A-frag (16x32 bf16): lane l -> row = l&15,
    //   k_local = ((l&16)?16:0) + i
    for (int p = tid; p < NKT * 512; p += 256) {
        int i  = p & 15;
        int l  = (p >> 4) & 31;
        int kt = p >> 9;
        int row = l & 15;
        int kl  = ((l & 16) ? 16 : 0) + i;
        int k   = kt * 32 + kl;
        int n   = n0 + row;
        long src;
        if (WINDOWED) {
            int c = k >> 3, kk = k & 7;
            src = ((long)n * Cc + c) * Tpad + (t + kk);
        } else {
            src = ((long)t * Nrow + n) * H2 + k;
        }
        sAhp[p] = Ahi[src];
        sAlp[p] = Alo[src];
    }
    __syncthreads();

    const int wave = tid >> 5, lane = tid & 31;
    const int ct0 = wave * 2, ct1 = wave * 2 + 1;
    v8f acc0, acc1;
    #pragma unroll
    for (int e = 0; e < 8; ++e) { acc0[e] = 0.f; acc1[e] = 0.f; }

    const v16bf* Bh = (const v16bf*)Bhi;
    const v16bf* Bl = (const v16bf*)Blo;

    for (int kt = 0; kt < NKT; ++kt) {
        v16bf ah = sAh[kt * 32 + lane];
        v16bf al = sAl[kt * 32 + lane];
        v16bf bh0 = Bh[(kt * 16 + ct0) * 32 + lane];
        v16bf bl0 = Bl[(kt * 16 + ct0) * 32 + lane];
        v16bf bh1 = Bh[(kt * 16 + ct1) * 32 + lane];
        v16bf bl1 = Bl[(kt * 16 + ct1) * 32 + lane];
        acc0 = wmma_bf16(ah, bh0, acc0);
        acc0 = wmma_bf16(ah, bl0, acc0);
        acc0 = wmma_bf16(al, bh0, acc0);
        acc1 = wmma_bf16(ah, bh1, acc1);
        acc1 = wmma_bf16(ah, bl1, acc1);
        acc1 = wmma_bf16(al, bh1, acc1);
    }

    // D-frag store: VGPR e -> M = (lane<16 ? e : e+8), N = lane&15
    const int mbase = (lane & 16) ? 8 : 0;
    const int c0 = ct0 * 16 + (lane & 15);
    const int c1 = ct1 * 16 + (lane & 15);
    #pragma unroll
    for (int e = 0; e < 8; ++e) {
        long o = ((long)t * Nrow + n0 + mbase + e) * 256;
        U[o + c0] = acc0[e];
        U[o + c1] = acc1[e];
    }
}

// ---------------------------------------------------------------------------
// Stage 4: SRU bidirectional scan. One thread per (dir, n, h-lane).
// ---------------------------------------------------------------------------
__global__ __launch_bounds__(256)
void scan_kernel(const float* __restrict__ U,
                 const float* __restrict__ v, const float* __restrict__ bprm,
                 int layer,
                 float* __restrict__ hf32,
                 unsigned short* __restrict__ hhi, unsigned short* __restrict__ hlo) {
    int g = blockIdx.x * 256 + threadIdx.x;   // 0..16383
    int j   = g & 31;
    int n   = (g >> 5) & 255;
    int dir = g >> 13;
    const float* vp = v    + (layer * 2 + dir) * H2;
    const float* bp = bprm + (layer * 2 + dir) * H2;
    float vf = vp[j], vr = vp[32 + j];
    float bf = bp[j], br = bp[32 + j];
    float c = 0.f;
    for (int tt = 0; tt < Tn; ++tt) {
        int t = dir ? (Tn - 1 - tt) : tt;
        long base = ((long)t * Nrow + n) * 256 + dir * H4 + j;
        float z  = U[base];
        float fp = U[base + 32];
        float rp = U[base + 64];
        float xp = U[base + 96];
        float f = 1.f / (1.f + __expf(-(fp + vf * c + bf)));
        float r = 1.f / (1.f + __expf(-(rp + vr * c + br)));
        c = f * c + (1.f - f) * z;
        float h = r * c + (1.f - r) * xp;
        long o = ((long)t * Nrow + n) * H2 + dir * Hh + j;
        hf32[o] = h;
        unsigned short hb = f2bf(h);
        hhi[o] = hb;
        hlo[o] = f2bf(h - bf2f(hb));
    }
}

// ---------------------------------------------------------------------------
// Stage 5: causal conv (K=8, 64->64) + bias + residual + output transpose.
//   out[b,c,f,t] = x[b,c,f,t] + bct[c] + sum_{m,i} h[t-m][n][i] * wt[c][m][i]
// One block per n = b*128+f; stage the h-slice (243x64 f32 = 61KB) in LDS.
// ---------------------------------------------------------------------------
__global__ __launch_bounds__(256)
void conv_kernel(const float* __restrict__ h, const float* __restrict__ wt,
                 const float* __restrict__ bct, const float* __restrict__ x,
                 float* __restrict__ out) {
    __shared__ float hs[Tn * H2];   // 62,208 bytes
    int n = blockIdx.x;
    int tid = threadIdx.x;
    for (int p = tid; p < Tn * H2; p += 256) {
        int t = p >> 6, i = p & 63;
        hs[p] = h[((long)t * Nrow + n) * H2 + i];
    }
    __syncthreads();
    int bat = n >> 7, f = n & 127;
    for (int idx = tid; idx < Cc * Tdim; idx += 256) {
        int c = idx / Tdim;
        int t = idx - c * Tdim;
        float acc = bct[c];
        int mlo = (t > Tn - 1) ? (t - (Tn - 1)) : 0;
        int mhi = (t < Kw - 1) ? t : (Kw - 1);
        for (int m = mlo; m <= mhi; ++m) {
            const float* hr = &hs[(t - m) * H2];
            const float* wr = &wt[(c * Kw + m) * H2];
            float a = 0.f;
            #pragma unroll
            for (int i2 = 0; i2 < H2; ++i2) a += hr[i2] * wr[i2];
            acc += a;
        }
        long o = (((long)(bat * Cc + c)) * Fdim + f) * Tdim + t;
        out[o] = acc + x[o];
    }
}

// ---------------------------------------------------------------------------
extern "C" void kernel_launch(void* const* d_in, const int* in_sizes, int n_in,
                              void* d_out, int out_size, void* d_ws, size_t ws_size,
                              hipStream_t stream) {
    const float* x     = (const float*)d_in[0];
    const float* gamma = (const float*)d_in[1];
    const float* beta  = (const float*)d_in[2];
    const float* W0    = (const float*)d_in[3];
    const float* Wr    = (const float*)d_in[4];
    const float* v     = (const float*)d_in[5];
    const float* b     = (const float*)d_in[6];
    const float* wct   = (const float*)d_in[7];
    const float* bct   = (const float*)d_in[8];
    float* out = (float*)d_out;

    char* ws = (char*)d_ws;
    size_t off = 0;
    auto alloc = [&](size_t bytes) -> void* {
        void* p = ws + off;
        off += (bytes + 255) & ~(size_t)255;
        return p;
    };

    float*          part  = (float*)alloc(512 * sizeof(float));
    float*          stats = (float*)alloc(16);
    unsigned short* xhi   = (unsigned short*)alloc((size_t)Nrow * Cc * Tpad * 2);
    unsigned short* xlo   = (unsigned short*)alloc((size_t)Nrow * Cc * Tpad * 2);
    unsigned short* pW0h  = (unsigned short*)alloc(131072 * 2);
    unsigned short* pW0l  = (unsigned short*)alloc(131072 * 2);
    unsigned short* pWrh  = (unsigned short*)alloc(49152 * 2);
    unsigned short* pWrl  = (unsigned short*)alloc(49152 * 2);
    float*          wt    = (float*)alloc(32768 * sizeof(float));
    float*          U     = (float*)alloc((size_t)Tn * Nrow * 256 * sizeof(float));
    float*          hf32  = (float*)alloc((size_t)Tn * Nrow * H2 * sizeof(float));
    unsigned short* hhi   = (unsigned short*)alloc((size_t)Tn * Nrow * H2 * 2);
    unsigned short* hlo   = (unsigned short*)alloc((size_t)Tn * Nrow * H2 * 2);

    stats_part_kernel<<<256, 256, 0, stream>>>(x, part);
    stats_final_kernel<<<1, 256, 0, stream>>>(part, stats);

    const int totalX = Nrow * Cc * Tdim;
    pack_x_kernel<<<(totalX + 255) / 256, 256, 0, stream>>>(x, gamma, beta, stats, xhi, xlo);
    pack_w_kernel<<<(131072 + 49152 + 32768) / 256, 256, 0, stream>>>(
        W0, Wr, wct, pW0h, pW0l, pWrh, pWrl, wt);

    // Layer 0: windowed GEMM (Ktot = 512 -> 16 k-tiles)
    gemm_kernel<16, true><<<Tn * 16, 256, 0, stream>>>(xhi, xlo, pW0h, pW0l, U);
    scan_kernel<<<64, 256, 0, stream>>>(U, v, b, 0, hf32, hhi, hlo);

    // Layers 1..3: recurrent GEMM (Ktot = 64 -> 2 k-tiles)
    for (int l = 1; l < 4; ++l) {
        gemm_kernel<2, false><<<Tn * 16, 256, 0, stream>>>(
            hhi, hlo, pWrh + (l - 1) * 16384, pWrl + (l - 1) * 16384, U);
        scan_kernel<<<64, 256, 0, stream>>>(U, v, b, l, hf32, hhi, hlo);
    }

    conv_kernel<<<Nrow, 256, 0, stream>>>(hf32, wt, bct, x, out);
}
